// EdgeLoss_5050881540109
// MI455X (gfx1250) — compile-verified, matrix operand order checked by reading, not run on hardware
//
#include <hip/hip_runtime.h>

typedef __attribute__((ext_vector_type(16))) _Float16 v16h;
typedef __attribute__((ext_vector_type(8)))  float    v8f;

#define HH 512
#define WW 512
#define HW (HH * WW)
#define TILE 32
#define HALO (TILE + 2)

// One block computes a 32x32 output tile for one batch image:
//  - load 34x34 grayscale halo (replicate-clamped) for pred & target into LDS
//  - 3x3 Sobel from LDS, magnitude, |pe - te| accumulated per thread
//  - block reduction: LDS -> wave0 packs 256 f32 partials as hi/lo f16
//    pairs into WMMA A, all-ones B, f32 accum -> v_wmma_f32_16x16x32_f16
__global__ __launch_bounds__(256) void edge_tile_kernel(
    const float* __restrict__ pred, const float* __restrict__ targ,
    float* __restrict__ partial)
{
  __shared__ float sp[HALO][HALO];
  __shared__ float st[HALO][HALO];
  __shared__ float red[256];

  const int tid = threadIdx.x;
  const int x0 = blockIdx.x * TILE;
  const int y0 = blockIdx.y * TILE;
  const int b  = blockIdx.z;
  const float* __restrict__ P = pred + (size_t)b * 3 * HW;
  const float* __restrict__ T = targ + (size_t)b * 3 * HW;

  // Grayscale-on-load with replicate padding (coordinate clamp).
  for (int i = tid; i < HALO * HALO; i += 256) {
    const int ly = i / HALO;
    const int lx = i - ly * HALO;
    int gy = y0 + ly - 1; gy = gy < 0 ? 0 : (gy > HH - 1 ? HH - 1 : gy);
    int gx = x0 + lx - 1; gx = gx < 0 ? 0 : (gx > WW - 1 ? WW - 1 : gx);
    const int idx = gy * WW + gx;
    const float pr = P[idx], pg = P[HW + idx], pb = P[2 * HW + idx];
    const float tr = T[idx], tg = T[HW + idx], tb = T[2 * HW + idx];
    sp[ly][lx] = 0.299f * pr + 0.587f * pg + 0.114f * pb;
    st[ly][lx] = 0.299f * tr + 0.587f * tg + 0.114f * tb;
  }
  __syncthreads();

  // Each thread: 4 pixels (rows tid>>5 + {0,8,16,24}, col tid&31).
  float acc = 0.0f;
  const int lx  = tid & 31;
  const int ly0 = tid >> 5;
#pragma unroll
  for (int q = 0; q < 4; ++q) {
    const int cy = ly0 + q * 8 + 1;
    const int cx = lx + 1;

    const float a00 = sp[cy-1][cx-1], a01 = sp[cy-1][cx], a02 = sp[cy-1][cx+1];
    const float a10 = sp[cy  ][cx-1],                     a12 = sp[cy  ][cx+1];
    const float a20 = sp[cy+1][cx-1], a21 = sp[cy+1][cx], a22 = sp[cy+1][cx+1];
    const float pgx = (a02 - a00) + 2.0f * (a12 - a10) + (a22 - a20);
    const float pgy = (a20 - a00) + 2.0f * (a21 - a01) + (a22 - a02);
    const float magp = sqrtf((pgx * pgx + pgy * pgy) * (1.0f / 64.0f) + 1e-6f);

    const float b00 = st[cy-1][cx-1], b01 = st[cy-1][cx], b02 = st[cy-1][cx+1];
    const float b10 = st[cy  ][cx-1],                     b12 = st[cy  ][cx+1];
    const float b20 = st[cy+1][cx-1], b21 = st[cy+1][cx], b22 = st[cy+1][cx+1];
    const float tgx = (b02 - b00) + 2.0f * (b12 - b10) + (b22 - b20);
    const float tgy = (b20 - b00) + 2.0f * (b21 - b01) + (b22 - b02);
    const float magt = sqrtf((tgx * tgx + tgy * tgy) * (1.0f / 64.0f) + 1e-6f);

    acc += fabsf(magp - magt);
  }
  red[tid] = acc;
  __syncthreads();

  // Wave 0 (32 lanes, EXEC all-1s inside this branch) reduces 256 partials
  // with a single WMMA: lane L packs red[8L..8L+7] as f16 hi/lo pairs.
  if (tid < 32) {
    v16h a;
#pragma unroll
    for (int k = 0; k < 8; ++k) {
      const float x  = red[tid * 8 + k];
      const _Float16 hi = (_Float16)x;
      a[k]     = hi;                      // high part
      a[k + 8] = (_Float16)(x - (float)hi); // residual, rounded to f16: hi+lo ~= x
    }
    v16h ones;
#pragma unroll
    for (int k = 0; k < 16; ++k) ones[k] = (_Float16)1.0f;

    v8f c = {};
    // D[m][n] = sum_k A[m][k]  (B all ones, f32 accumulate)
    c = __builtin_amdgcn_wmma_f32_16x16x32_f16(
        /*neg_a=*/false, a, /*neg_b=*/false, ones,
        /*c_mod=*/(short)0, c, /*reuse_a=*/false, /*reuse_b=*/false);

    // Lanes 0-15 hold rows 0-7 of D in c[0..7]; lanes 16-31 hold rows 8-15.
    float s = c[0] + c[1] + c[2] + c[3] + c[4] + c[5] + c[6] + c[7];
    s += __shfl_xor(s, 16, 32);   // combine the two half-waves (wave32)

    if (tid == 0) {
      const int pid = ((int)blockIdx.z * (int)gridDim.y + (int)blockIdx.y) *
                          (int)gridDim.x + (int)blockIdx.x;
      partial[pid] = s;
    }
  }
}

// Deterministic fixed-order final reduction of per-block partials -> mean.
__global__ __launch_bounds__(256) void final_reduce_kernel(
    const float* __restrict__ partial, float* __restrict__ out,
    int n, float invN)
{
  __shared__ float s[256];
  float a = 0.0f;
  for (int i = threadIdx.x; i < n; i += 256) a += partial[i];
  s[threadIdx.x] = a;
  __syncthreads();
#pragma unroll
  for (int stp = 128; stp > 0; stp >>= 1) {
    if ((int)threadIdx.x < stp) s[threadIdx.x] += s[threadIdx.x + stp];
    __syncthreads();
  }
  if (threadIdx.x == 0) out[0] = s[0] * invN;
}

extern "C" void kernel_launch(void* const* d_in, const int* in_sizes, int n_in,
                              void* d_out, int out_size, void* d_ws, size_t ws_size,
                              hipStream_t stream) {
  (void)n_in; (void)out_size; (void)ws_size;
  const float* pred = (const float*)d_in[0];
  const float* targ = (const float*)d_in[1];
  float* partial = (float*)d_ws;               // nblocks floats of scratch

  const int B = in_sizes[0] / (3 * HW);        // batch (16 for the reference)
  dim3 grid(WW / TILE, HH / TILE, B);          // 16 x 16 x B = 4096 blocks

  edge_tile_kernel<<<grid, 256, 0, stream>>>(pred, targ, partial);

  const int nblocks = (WW / TILE) * (HH / TILE) * B;
  const float invN = 1.0f / ((float)B * (float)HW);
  final_reduce_kernel<<<1, 256, 0, stream>>>(partial, (float*)d_out, nblocks, invN);
}
